// MultiLayerRNN_24902220382918
// MI455X (gfx1250) — matrix operand order repaired and checked by reading, not compiled, not run on hardware
//
#include <hip/hip_runtime.h>

// Problem constants (from reference)
#define B_  16
#define T_  256
#define H_  1024
#define E_  256
#define V_  32000
#define M_  (B_ * T_)   // 4096 rows (b*T + t)

// GEMM block tiling
#define BM 128
#define BN 128
#define BK 64

typedef unsigned int       u32;
typedef unsigned long long u64;
typedef __attribute__((ext_vector_type(16))) __bf16 v16bf;
typedef __attribute__((ext_vector_type(8)))  float  v8f;
typedef __attribute__((ext_vector_type(4)))  u32    u32x4;
typedef __attribute__((ext_vector_type(8)))  int    i32x8;
typedef __attribute__((ext_vector_type(4)))  int    i32x4;

union Frag16 { v16bf v; unsigned d[8]; };

#if defined(__has_builtin)
#if __has_builtin(__builtin_amdgcn_tensor_load_to_lds) && __has_builtin(__builtin_amdgcn_s_wait_tensorcnt)
#define HAVE_TDM 1
#endif
#endif
#ifndef HAVE_TDM
#define HAVE_TDM 0
#endif

__device__ __forceinline__ __bf16 f2bf(float f) {
  unsigned u = __builtin_bit_cast(unsigned, f);
  unsigned r = (u + 0x7FFFu + ((u >> 16) & 1u)) >> 16;   // round-to-nearest-even
  unsigned short s = (unsigned short)r;
  return __builtin_bit_cast(__bf16, s);
}

__device__ __forceinline__ float sigmoidf_(float x) {
  return 1.0f / (1.0f + __expf(-x));
}

#if HAVE_TDM
// ---------------------------------------------------------------------------
// Issue one TDM 2D tile load: global (row-major, row_stride elems of 2B)
// -> packed LDS tile [tile_d1][tile_d0] bf16. Descriptor per ISA ch.8.
// Must be executed by exactly one wave (EXEC ignored; 1 issue per wave).
// Toolchain here is the clang-23 lane: 6-arg builtin
// (uint32x4 g0, int32x8 g1, int32x4 g2, int32x4 g3, int32x8 extra, i32 cpol).
// ---------------------------------------------------------------------------
__device__ __forceinline__ void tdm_load_2d(u32 lds_addr, const void* gaddr,
                                            u32 tensor_d0, u32 tensor_d1,
                                            u32 tile_d0, u32 tile_d1,
                                            u64 row_stride) {
  u64 ga = (u64)(uintptr_t)gaddr;
  u32x4 g0;
  g0.x = 0x1u;                                            // count=1, load, no gather
  g0.y = lds_addr;                                        // LDS byte address
  g0.z = (u32)ga;                                         // global_addr[31:0]
  g0.w = ((u32)(ga >> 32) & 0x1FFFFFFu) | (2u << 30);     // addr[56:32] | type=2
  i32x8 g1;
  g1[0] = (int)0x00010000u;                               // data_size=1 (2 bytes/elem)
  g1[1] = (int)((tensor_d0 & 0xFFFFu) << 16);             // tensor_dim0[15:0] @ bits63:48
  g1[2] = (int)((tensor_d0 >> 16) | ((tensor_d1 & 0xFFFFu) << 16));
  g1[3] = (int)((tensor_d1 >> 16) | (tile_d0 << 16));     // tile_dim0 @ bits127:112
  g1[4] = (int)(tile_d1 & 0xFFFFu);                       // tile_dim1; tile_dim2=0
  g1[5] = (int)(u32)(row_stride & 0xFFFFFFFFu);           // tensor_dim0_stride lo32
  g1[6] = (int)((u32)(row_stride >> 32) & 0xFFFFu);       // stride hi16; dim1_stride=0
  g1[7] = 0;
  i32x4 gz  = {0, 0, 0, 0};                               // groups 2/3 unused (2D)
  i32x8 gz8 = {0, 0, 0, 0, 0, 0, 0, 0};
  __builtin_amdgcn_tensor_load_to_lds(g0, g1, gz, gz, gz8, 0);
}
#endif

// ---------------------------------------------------------------------------
// f32 -> bf16 elementwise convert
// ---------------------------------------------------------------------------
__global__ __launch_bounds__(256)
void f32_to_bf16_kern(const float* __restrict__ in, __bf16* __restrict__ out, size_t n) {
  size_t i = (size_t)blockIdx.x * blockDim.x + threadIdx.x;
  if (i < n) out[i] = f2bf(in[i]);
}

// ---------------------------------------------------------------------------
// Embedding gather: xs_bf[row, e] = bf16(emb[x[row], e]),  row = b*T + t
// ---------------------------------------------------------------------------
__global__ __launch_bounds__(256)
void embed_gather_kern(const int* __restrict__ x, const float* __restrict__ emb,
                       __bf16* __restrict__ xs) {
  int row = blockIdx.x;
  int e   = threadIdx.x;
  int tok = x[row];
  xs[(size_t)row * E_ + e] = f2bf(emb[(size_t)tok * E_ + e]);
}

// ---------------------------------------------------------------------------
// bf16 WMMA GEMM:  out[M,N] = A[M,K] * W[N,K]^T (+ bias[n])
// Block: 256 threads = 8 waves (2x4). Block tile 128x128, K-slab 64,
// double-buffered LDS staged by the Tensor Data Mover (wave 0 issues,
// s_wait_tensorcnt + barrier gates consumers). Wave tile 64x32 (4x2 WMMA).
// All dims divide tiles evenly for this problem.
// If out_last != nullptr, rows with (m % 256 == 255) also write
// out_last[(m/256)*N + n]  (the reference's out[:, -1]).
// ---------------------------------------------------------------------------
__global__ __launch_bounds__(256)
void gemm_bf16_wmma(const __bf16* __restrict__ A, const __bf16* __restrict__ W,
                    const float* __restrict__ bias, float* __restrict__ out,
                    float* __restrict__ out_last, int M, int N, int K) {
  __shared__ __bf16 aTile[2][BM * BK];   // 2 x 16 KB
  __shared__ __bf16 bTile[2][BN * BK];   // 2 x 16 KB
  const int lane = threadIdx.x & 31;
  const int l15  = lane & 15;
  const int kh   = lane >> 4;
  const int wave = threadIdx.x >> 5;
  const int wm   = wave >> 2;           // 0..1
  const int wn   = wave & 3;            // 0..3
  const int m0g  = blockIdx.y * BM;
  const int n0g  = blockIdx.x * BN;
  const int wmo  = wm * 64;             // wave M offset in block tile
  const int wno  = wn * 32;             // wave N offset in block tile

  v8f acc[4][2] = {};
  const int S = K / BK;

#if HAVE_TDM
  if (wave == 0) {
    tdm_load_2d((u32)(uintptr_t)&aTile[0][0], A + (size_t)m0g * K,
                (u32)K, (u32)M, BK, BM, (u64)K);
    tdm_load_2d((u32)(uintptr_t)&bTile[0][0], W + (size_t)n0g * K,
                (u32)K, (u32)N, BK, BN, (u64)K);
  }
#endif

  for (int s = 0; s < S; ++s) {
    const int cur = s & 1;
#if HAVE_TDM
    const int nxt = cur ^ 1;
    if (wave == 0) {
      if (s + 1 < S) {
        const int kb = (s + 1) * BK;
        tdm_load_2d((u32)(uintptr_t)&aTile[nxt][0], A + (size_t)m0g * K + kb,
                    (u32)K, (u32)M, BK, BM, (u64)K);
        tdm_load_2d((u32)(uintptr_t)&bTile[nxt][0], W + (size_t)n0g * K + kb,
                    (u32)K, (u32)N, BK, BN, (u64)K);
        __builtin_amdgcn_s_wait_tensorcnt((short)2);   // slab s's pair complete
      } else {
        __builtin_amdgcn_s_wait_tensorcnt((short)0);
      }
    }
    __syncthreads();
#else
    {
      // Fallback cooperative copy (no TDM builtin on this toolchain)
      const int kb = s * BK;
      const uint4* __restrict__ ag = (const uint4*)(A + (size_t)m0g * K + kb);
      const uint4* __restrict__ bg = (const uint4*)(W + (size_t)n0g * K + kb);
      uint4* al = (uint4*)&aTile[cur][0];
      uint4* bl = (uint4*)&bTile[cur][0];
      const int rowStride16 = K >> 3;          // uint4 per global row
#pragma unroll
      for (int q = 0; q < 4; ++q) {
        int idx = (int)threadIdx.x + 256 * q;  // 1024 uint4 per tile
        int r = idx >> 3, c = idx & 7;
        al[idx] = ag[(size_t)r * rowStride16 + c];
        bl[idx] = bg[(size_t)r * rowStride16 + c];
      }
      __syncthreads();
    }
#endif

    const __bf16* at = &aTile[cur][0];
    const __bf16* bt = &bTile[cur][0];
#pragma unroll
    for (int k2 = 0; k2 < BK; k2 += 32) {
      Frag16 af[4], bf[2];
#pragma unroll
      for (int i = 0; i < 4; ++i) {
        const unsigned* ap = (const unsigned*)(at + (size_t)(wmo + i * 16 + l15) * BK + k2);
#pragma unroll
        for (int v = 0; v < 8; ++v)
          af[i].d[v] = ap[(v & 3) + 4 * kh + 8 * (v >> 2)];   // A 16x32 layout
      }
#pragma unroll
      for (int j = 0; j < 2; ++j) {
        const unsigned* bp = (const unsigned*)(bt + (size_t)(wno + j * 16 + l15) * BK + k2);
#pragma unroll
        for (int v = 0; v < 8; ++v)
          bf[j].d[v] = bp[(v & 3) + 4 * kh + 8 * (v >> 2)];   // B 32x16 layout
      }
#pragma unroll
      for (int i = 0; i < 4; ++i)
#pragma unroll
        for (int j = 0; j < 2; ++j)
          acc[i][j] = __builtin_amdgcn_wmma_f32_16x16x32_bf16(
              false, af[i].v, false, bf[j].v, (short)0, acc[i][j], false, false);
    }
    __syncthreads();
  }

#pragma unroll
  for (int i = 0; i < 4; ++i) {
#pragma unroll
    for (int j = 0; j < 2; ++j) {
      int gn = n0g + wno + j * 16 + l15;
      float bv = (bias != nullptr) ? bias[gn] : 0.0f;
#pragma unroll
      for (int v = 0; v < 8; ++v) {
        int gm = m0g + wmo + i * 16 + v + 8 * kh;   // C/D: row = vgpr + 8*(lane>=16)
        float val = acc[i][j][v] + bv;
        out[(size_t)gm * N + gn] = val;
        if (out_last != nullptr && ((gm & (T_ - 1)) == (T_ - 1)))
          out_last[(size_t)(gm >> 8) * N + gn] = val;
      }
    }
  }
}

// ---------------------------------------------------------------------------
// LSTM recurrence for one layer, single workgroup (global serial dependency
// on c each step). 512 threads = 16 waves; each wave owns 4 N-tiles of 16.
// c kept in f32 registers; bf16 mirror in LDS is the WMMA A operand.
// ---------------------------------------------------------------------------
__global__ __launch_bounds__(512)
void lstm_recur_kern(const float* __restrict__ gF, const float* __restrict__ gI,
                     const float* __restrict__ gO, const float* __restrict__ gC,
                     const float* __restrict__ bF, const float* __restrict__ bI,
                     const float* __restrict__ bO,
                     const __bf16* __restrict__ WF, const __bf16* __restrict__ WI,
                     const __bf16* __restrict__ WO,
                     const float* __restrict__ c0, __bf16* __restrict__ hOut) {
  __shared__ __bf16 c_sh[B_ * H_];               // 32 KB bf16 mirror of c [16,1024]
  const int lane = threadIdx.x & 31;
  const int l15  = lane & 15;
  const int kh   = lane >> 4;
  const int wave = threadIdx.x >> 5;             // 0..15

  float creg[4][8];
  float biasF[4], biasI[4], biasO[4];
#pragma unroll
  for (int j = 0; j < 4; ++j) {
    int n = (wave * 4 + j) * 16 + l15;
    float cv = c0[n];
    biasF[j] = bF[n]; biasI[j] = bI[n]; biasO[j] = bO[n];
#pragma unroll
    for (int v = 0; v < 8; ++v) {
      creg[j][v] = cv;
      c_sh[(size_t)(v + 8 * kh) * H_ + n] = f2bf(cv);
    }
  }
  __syncthreads();

  for (int t = 0; t < T_; ++t) {
    v8f aF[4], aI[4], aO[4];
    v8f z = {};
#pragma unroll
    for (int j = 0; j < 4; ++j) { aF[j] = z; aI[j] = z; aO[j] = z; }

    for (int kb = 0; kb < H_; kb += 32) {
      Frag16 a;
      const unsigned* ap = (const unsigned*)(c_sh + (size_t)l15 * H_ + kb);
#pragma unroll
      for (int v = 0; v < 8; ++v)
        a.d[v] = ap[(v & 3) + 4 * kh + 8 * (v >> 2)];

#pragma unroll
      for (int j = 0; j < 4; ++j) {
        int nrow = (wave * 4 + j) * 16 + l15;
        const unsigned* pf = (const unsigned*)(WF + (size_t)nrow * H_ + kb);
        const unsigned* pi = (const unsigned*)(WI + (size_t)nrow * H_ + kb);
        const unsigned* po = (const unsigned*)(WO + (size_t)nrow * H_ + kb);
        Frag16 fb, ib, ob;
#pragma unroll
        for (int v = 0; v < 8; ++v) {
          int bd = (v & 3) + 4 * kh + 8 * (v >> 2);
          fb.d[v] = pf[bd]; ib.d[v] = pi[bd]; ob.d[v] = po[bd];
        }
        aF[j] = __builtin_amdgcn_wmma_f32_16x16x32_bf16(false, a.v, false, fb.v, (short)0, aF[j], false, false);
        aI[j] = __builtin_amdgcn_wmma_f32_16x16x32_bf16(false, a.v, false, ib.v, (short)0, aI[j], false, false);
        aO[j] = __builtin_amdgcn_wmma_f32_16x16x32_bf16(false, a.v, false, ob.v, (short)0, aO[j], false, false);
      }
    }

#pragma unroll
    for (int j = 0; j < 4; ++j) {
      int n = (wave * 4 + j) * 16 + l15;
#pragma unroll
      for (int v = 0; v < 8; ++v) {
        int b = v + 8 * kh;
        size_t g = ((size_t)b * T_ + t) * H_ + n;
        float f  = sigmoidf_(aF[j][v] + gF[g] + biasF[j]);
        float i  = sigmoidf_(aI[j][v] + gI[g] + biasI[j]);
        float o  = sigmoidf_(aO[j][v] + gO[g] + biasO[j]);
        float cn = f * creg[j][v] + i * tanhf(gC[g]);
        creg[j][v] = cn;
        hOut[g] = f2bf(o * tanhf(cn));
      }
    }
    __syncthreads();
#pragma unroll
    for (int j = 0; j < 4; ++j) {
      int n = (wave * 4 + j) * 16 + l15;
#pragma unroll
      for (int v = 0; v < 8; ++v)
        c_sh[(size_t)(v + 8 * kh) * H_ + n] = f2bf(creg[j][v]);
    }
    __syncthreads();
  }
}

// ---------------------------------------------------------------------------
// Host-side orchestration
// ---------------------------------------------------------------------------
static inline void cvt(const float* src, __bf16* dst, size_t n, hipStream_t s) {
  unsigned blocks = (unsigned)((n + 255) / 256);
  f32_to_bf16_kern<<<blocks, 256, 0, s>>>(src, dst, n);
}

extern "C" void kernel_launch(void* const* d_in, const int* in_sizes, int n_in,
                              void* d_out, int out_size, void* d_ws, size_t ws_size,
                              hipStream_t stream) {
  (void)in_sizes; (void)n_in; (void)out_size; (void)ws_size;

  // ---- input pointers (recursive insertion-order flatten of setup_inputs) ----
  const int*   x   = (const int*)  d_in[0];
  const float* emb = (const float*)d_in[1];
  // per-layer dict order: Wxf, Wci, bci, Wxi, Wcf, bcf, Wxo, Wco, bco, Wxc, bxc, c0
  const float* Wx[2][4];   // f, i, o, c  (x-projections)
  const float* Wc[2][3];   // Wcf, Wci, Wco
  const float* bc[2][3];   // bcf, bci, bco
  const float* bxc[2];
  const float* c0p[2];
  for (int l = 0; l < 2; ++l) {
    int base = (l == 0) ? 2 : 14;
    Wx[l][0] = (const float*)d_in[base + 0];   // Wxf
    Wc[l][1] = (const float*)d_in[base + 1];   // Wci
    bc[l][1] = (const float*)d_in[base + 2];   // bci
    Wx[l][1] = (const float*)d_in[base + 3];   // Wxi
    Wc[l][0] = (const float*)d_in[base + 4];   // Wcf
    bc[l][0] = (const float*)d_in[base + 5];   // bcf
    Wx[l][2] = (const float*)d_in[base + 6];   // Wxo
    Wc[l][2] = (const float*)d_in[base + 7];   // Wco
    bc[l][2] = (const float*)d_in[base + 8];   // bco
    Wx[l][3] = (const float*)d_in[base + 9];   // Wxc
    bxc[l]   = (const float*)d_in[base + 10];  // bxc
    c0p[l]   = (const float*)d_in[base + 11];  // c0
  }
  const float* head_w = (const float*)d_in[26];
  const float* head_b = (const float*)d_in[27];

  // ---- workspace carve ----
  char* p = (char*)d_ws;
  auto carve = [&](size_t bytes) -> char* {
    char* r = p;
    p += (bytes + 255) & ~(size_t)255;
    return r;
  };
  __bf16* xs_bf = (__bf16*)carve((size_t)M_ * E_ * 2);
  __bf16* h1_bf = (__bf16*)carve((size_t)M_ * H_ * 2);
  __bf16* h2_bf = (__bf16*)carve((size_t)M_ * H_ * 2);
  float*  gFbuf = (float*) carve((size_t)M_ * H_ * 4);
  float*  gIbuf = (float*) carve((size_t)M_ * H_ * 4);
  float*  gObuf = (float*) carve((size_t)M_ * H_ * 4);
  float*  gCbuf = (float*) carve((size_t)M_ * H_ * 4);
  __bf16* Wx_bf[2][4];
  __bf16* Wc_bf[2][3];
  for (int l = 0; l < 2; ++l) {
    size_t inF = (l == 0) ? E_ : H_;
    for (int g = 0; g < 4; ++g) Wx_bf[l][g] = (__bf16*)carve((size_t)H_ * inF * 2);
    for (int g = 0; g < 3; ++g) Wc_bf[l][g] = (__bf16*)carve((size_t)H_ * H_ * 2);
  }
  __bf16* headw_bf = (__bf16*)carve((size_t)V_ * H_ * 2);

  // ---- weight conversions (bf16 feeds WMMA) ----
  for (int l = 0; l < 2; ++l) {
    size_t inF = (l == 0) ? E_ : H_;
    for (int g = 0; g < 4; ++g) cvt(Wx[l][g], Wx_bf[l][g], (size_t)H_ * inF, stream);
    for (int g = 0; g < 3; ++g) cvt(Wc[l][g], Wc_bf[l][g], (size_t)H_ * H_, stream);
  }
  cvt(head_w, headw_bf, (size_t)V_ * H_, stream);

  // ---- embedding gather ----
  embed_gather_kern<<<M_, 256, 0, stream>>>(x, emb, xs_bf);

  // ---- per-layer: gate GEMMs then recurrence ----
  const __bf16* layer_in = xs_bf;
  __bf16* layer_out[2] = { h1_bf, h2_bf };
  for (int l = 0; l < 2; ++l) {
    int K = (l == 0) ? E_ : H_;
    dim3 grid(H_ / BN, M_ / BM);
    gemm_bf16_wmma<<<grid, 256, 0, stream>>>(layer_in, Wx_bf[l][0], nullptr, gFbuf, nullptr, M_, H_, K);
    gemm_bf16_wmma<<<grid, 256, 0, stream>>>(layer_in, Wx_bf[l][1], nullptr, gIbuf, nullptr, M_, H_, K);
    gemm_bf16_wmma<<<grid, 256, 0, stream>>>(layer_in, Wx_bf[l][2], nullptr, gObuf, nullptr, M_, H_, K);
    gemm_bf16_wmma<<<grid, 256, 0, stream>>>(layer_in, Wx_bf[l][3], bxc[l], gCbuf, nullptr, M_, H_, K);

    lstm_recur_kern<<<1, 512, 0, stream>>>(gFbuf, gIbuf, gObuf, gCbuf,
                                           bc[l][0], bc[l][1], bc[l][2],
                                           Wc_bf[l][0], Wc_bf[l][1], Wc_bf[l][2],
                                           c0p[l], layer_out[l]);
    layer_in = layer_out[l];
  }

  // ---- head GEMM (+bias, + out[:, -1] duplicate) ----
  float* out_full = (float*)d_out;
  float* out_last = out_full + (size_t)M_ * V_;
  dim3 hgrid(V_ / BN, M_ / BM);
  gemm_bf16_wmma<<<hgrid, 256, 0, stream>>>(h2_bf, headw_bf, head_b,
                                            out_full, out_last, M_, V_, H_);
}
